// VectorQuantizer_58454504898974
// MI455X (gfx1250) — matrix operand order repaired and checked by reading, not compiled, not run on hardware
//
#include <hip/hip_runtime.h>
#include <hip/hip_bf16.h>
#include <math.h>

// ---------------------------------------------------------------------------
// VQ layer for MI455X (gfx1250): bf16 WMMA (v_wmma_f32_16x16x32_bf16) GEMMs,
// async global->LDS B-tile staging (ASYNCcnt), fused argmax, codebook-gather.
// ---------------------------------------------------------------------------

#define BDIM   16
#define TDIM   2048
#define NTOK   (BDIM * TDIM)   // 32768
#define IN_DIM 1024
#define D_DIM  512
#define K_DIM  2048

typedef __bf16 bf16_t;
typedef float  v8f   __attribute__((ext_vector_type(8)));
typedef bf16_t v16bf __attribute__((ext_vector_type(16)));
typedef bf16_t v8bf  __attribute__((ext_vector_type(8)));
typedef bf16_t v4bf  __attribute__((ext_vector_type(4)));

// ---- CDNA5 async global->LDS copy (tracked by ASYNCcnt) --------------------
// Generic pointers to LDS lower as {aperture_hi32, lds_byte_offset}, so the
// low 32 bits of the flat address are the LDS address the instruction wants.
__device__ __forceinline__ void async_b128(const bf16_t* ldsDst, const bf16_t* gsrc) {
  unsigned lds = (unsigned)(unsigned long long)(uintptr_t)ldsDst;
  unsigned long long ga = (unsigned long long)(uintptr_t)gsrc;
  asm volatile("global_load_async_to_lds_b128 %0, %1, off"
               :: "v"(lds), "v"(ga) : "memory");
}
__device__ __forceinline__ void wait_async0() {
  asm volatile("s_wait_asynccnt 0x0" ::: "memory");
}

// ---------------- LDS fragment loaders (per CDNA5 ISA §7.12.2) --------------
// A (16x32 bf16): lanes 0-15 -> row M=lane, K elems {kk+0..7, kk+16..23}
//                 lanes 16-31 -> row M=lane-16, K elems {kk+8..15, kk+24..31}
__device__ __forceinline__ v16bf ldsA_frag(const bf16_t* As, int stride,
                                           int m0, int kk, int lane) {
  int r  = lane & 15;
  int kq = (lane >> 4) * 8;
  const bf16_t* p = As + (m0 + r) * stride + kk + kq;
  union { v8bf h[2]; v16bf v; } u;
  u.h[0] = *(const v8bf*)(p);
  u.h[1] = *(const v8bf*)(p + 16);
  return u.v;
}
// B (32x16 bf16, staged K-major per column): lanes 0-15 -> col N=lane,
// K = {kk+0..7, kk+16..23}; lanes 16-31 -> col N=lane-16, K = {kk+8..15, kk+24..31}.
__device__ __forceinline__ v16bf ldsB_frag(const bf16_t* Bs, int stride,
                                           int n0, int kk, int lane) {
  int r  = lane & 15;
  int kq = (lane >> 4) * 8;
  const bf16_t* p = Bs + (n0 + r) * stride + kk + kq;
  union { v8bf h[2]; v16bf v; } u;
  u.h[0] = *(const v8bf*)(p);
  u.h[1] = *(const v8bf*)(p + 16);
  return u.v;
}

#define BM   128
#define BN   128
#define BK   64
#define LDSK 72   // pad: 144B row stride -> conflict-free b128 LDS reads

// ------------------- Generic GEMM: C = A(f32) * Bt(bf16)^T + bias ----------
__global__ __launch_bounds__(256)
void vq_gemm_bf16_store(const float* __restrict__ A,
                        const bf16_t* __restrict__ Bt,
                        const float* __restrict__ bias,
                        float* __restrict__ C,
                        int M, int N, int Kd) {
  __shared__ __align__(16) bf16_t As[BM * LDSK];
  __shared__ __align__(16) bf16_t Bs[BN * LDSK];
  const int tid   = threadIdx.x;
  const int lane  = tid & 31;
  const int wave  = tid >> 5;
  const int waveM = (wave & 3) * 32;
  const int waveN = (wave >> 2) * 64;
  const int blockM = blockIdx.x * BM;
  const int blockN = blockIdx.y * BN;

  v8f acc[2][4];
#pragma unroll
  for (int i = 0; i < 2; ++i)
#pragma unroll
    for (int j = 0; j < 4; ++j)
      acc[i][j] = (v8f){0.f, 0.f, 0.f, 0.f, 0.f, 0.f, 0.f, 0.f};

  for (int k0 = 0; k0 < Kd; k0 += BK) {
    // B tile (128 n-rows x 64 k): pure bf16 copy -> async global->LDS, no VGPRs
    {
      int c8 = tid & 7;
      int r0 = tid >> 3;   // 0..31
#pragma unroll
      for (int p = 0; p < 4; ++p) {
        int r = r0 + p * 32;
        async_b128(&Bs[r * LDSK + c8 * 8],
                   Bt + (size_t)(blockN + r) * Kd + k0 + c8 * 8);
      }
    }
    // A tile (128 x 64) fp32 -> bf16, two-phase so all loads are in flight
    {
      int c4 = tid & 15;   // float4 column
      int r0 = tid >> 4;   // 0..15
      float4 tmp[8];
#pragma unroll
      for (int p = 0; p < 8; ++p)
        tmp[p] = *(const float4*)(A + (size_t)(blockM + r0 + p * 16) * Kd + k0 + c4 * 4);
#pragma unroll
      for (int p = 0; p < 8; ++p) {
        v4bf o;
        o[0] = (bf16_t)tmp[p].x; o[1] = (bf16_t)tmp[p].y;
        o[2] = (bf16_t)tmp[p].z; o[3] = (bf16_t)tmp[p].w;
        *(v4bf*)&As[(r0 + p * 16) * LDSK + c4 * 4] = o;
      }
    }
    wait_async0();
    __syncthreads();
#pragma unroll
    for (int kk = 0; kk < BK; kk += 32) {
      v16bf a0 = ldsA_frag(As, LDSK, waveM,      kk, lane);
      v16bf a1 = ldsA_frag(As, LDSK, waveM + 16, kk, lane);
#pragma unroll
      for (int j = 0; j < 4; ++j) {
        v16bf b = ldsB_frag(Bs, LDSK, waveN + j * 16, kk, lane);
        acc[0][j] = __builtin_amdgcn_wmma_f32_16x16x32_bf16(
            false, a0, false, b, (short)0, acc[0][j], false, false);
        acc[1][j] = __builtin_amdgcn_wmma_f32_16x16x32_bf16(
            false, a1, false, b, (short)0, acc[1][j], false, false);
      }
    }
    __syncthreads();
  }
  // epilogue: C layout — VGPR v: lanes0-15 M=v, lanes16-31 M=8+v; N=lane&15
  const int rhalf = (lane >> 4) * 8;
  const int cl = lane & 15;
#pragma unroll
  for (int i = 0; i < 2; ++i)
#pragma unroll
    for (int j = 0; j < 4; ++j) {
      int col = blockN + waveN + j * 16 + cl;
      float bv = bias ? bias[col] : 0.f;
#pragma unroll
      for (int v = 0; v < 8; ++v) {
        int row = blockM + waveM + i * 16 + rhalf + v;
        C[(size_t)row * N + col] = acc[i][j][v] + bv;
      }
    }
}

// ----------- GEMM2 + fused row argmax: idx[n] = argmax_k (x @ en_norm^T) ----
// A-strip (128 x 512 bf16) resident in LDS (CDNA5: 320KB LDS/WGP).
#define A2LD 552  // 512 + 40 pad: conflict-free b128 reads, 16B aligned rows
__global__ __launch_bounds__(256)
void vq_gemm_argmax(const float* __restrict__ A,    // x [32768][512]
                    const bf16_t* __restrict__ Bt,  // en_norm^T [2048][512]
                    int* __restrict__ outIdx) {
  __shared__ __align__(16) bf16_t As2[BM * A2LD];   // ~141 KB
  __shared__ __align__(16) bf16_t Bs[BN * LDSK];    // ~18 KB
  const int tid  = threadIdx.x;
  const int lane = tid & 31;
  const int wave = tid >> 5;        // 8 waves, 16 rows each
  const int blockM = blockIdx.x * BM;

  // stage full A strip once: 128 rows x 512 k, fp32 -> bf16, batched 8-deep
  for (int it = 0; it < 64; it += 8) {
    float4 tmp[8];
#pragma unroll
    for (int p = 0; p < 8; ++p) {
      int idx = tid + (it + p) * 256;
      int r = idx >> 7, c4 = idx & 127;
      tmp[p] = *(const float4*)(A + (size_t)(blockM + r) * D_DIM + c4 * 4);
    }
#pragma unroll
    for (int p = 0; p < 8; ++p) {
      int idx = tid + (it + p) * 256;
      int r = idx >> 7, c4 = idx & 127;
      v4bf o;
      o[0] = (bf16_t)tmp[p].x; o[1] = (bf16_t)tmp[p].y;
      o[2] = (bf16_t)tmp[p].z; o[3] = (bf16_t)tmp[p].w;
      *(v4bf*)&As2[r * A2LD + c4 * 4] = o;
    }
  }

  float bestVal[8];
  int   bestIdx[8];
#pragma unroll
  for (int v = 0; v < 8; ++v) { bestVal[v] = -INFINITY; bestIdx[v] = 0; }

  for (int chunk = 0; chunk < K_DIM / BN; ++chunk) {
    const int n0 = chunk * BN;
    v8f acc[8];
#pragma unroll
    for (int j = 0; j < 8; ++j)
      acc[j] = (v8f){0.f, 0.f, 0.f, 0.f, 0.f, 0.f, 0.f, 0.f};

    for (int k0 = 0; k0 < D_DIM; k0 += BK) {
      {
        int c8 = tid & 7;
        int r0 = tid >> 3;
#pragma unroll
        for (int p = 0; p < 4; ++p) {
          int r = r0 + p * 32;
          async_b128(&Bs[r * LDSK + c8 * 8],
                     Bt + (size_t)(n0 + r) * D_DIM + k0 + c8 * 8);
        }
      }
      wait_async0();
      __syncthreads();
#pragma unroll
      for (int kk = 0; kk < BK; kk += 32) {
        v16bf a = ldsA_frag(As2, A2LD, wave * 16, k0 + kk, lane);
#pragma unroll
        for (int j = 0; j < 8; ++j) {
          v16bf b = ldsB_frag(Bs, LDSK, j * 16, kk, lane);
          acc[j] = __builtin_amdgcn_wmma_f32_16x16x32_bf16(
              false, a, false, b, (short)0, acc[j], false, false);
        }
      }
      __syncthreads();
    }
    // in-register argmax merge. Row of acc[j][v]: wave*16 + (lane>>4)*8 + v,
    // col: n0 + j*16 + (lane&15). Reduce over the 16-lane half-group.
#pragma unroll
    for (int j = 0; j < 8; ++j) {
#pragma unroll
      for (int v = 0; v < 8; ++v) {
        float val = acc[j][v];
        int   col = n0 + j * 16 + (lane & 15);
#pragma unroll
        for (int off = 8; off > 0; off >>= 1) {
          float ov = __shfl_xor(val, off, 16);
          int   oc = __shfl_xor(col, off, 16);
          if (ov > val || (ov == val && oc < col)) { val = ov; col = oc; }
        }
        if (val > bestVal[v] || (val == bestVal[v] && col < bestIdx[v])) {
          bestVal[v] = val; bestIdx[v] = col;
        }
      }
    }
  }
  if ((lane & 15) == 0) {
    int g = lane >> 4;
#pragma unroll
    for (int v = 0; v < 8; ++v)
      outIdx[blockM + wave * 16 + g * 8 + v] = bestIdx[v];
  }
}

// ------------------------------- prep kernels ------------------------------
__global__ void vq_colnorm(const float* __restrict__ emb, float* __restrict__ norms) {
  int k = blockIdx.x * blockDim.x + threadIdx.x;
  if (k >= K_DIM) return;
  float s = 0.f;
  for (int d = 0; d < D_DIM; ++d) { float v = emb[(size_t)d * K_DIM + k]; s += v * v; }
  norms[k] = sqrtf(s);
}

__global__ void vq_prep_codebook(const float* __restrict__ emb,
                                 const float* __restrict__ norms,
                                 bf16_t* __restrict__ ent,     // [K][D] normalized bf16
                                 float* __restrict__ embT) {   // [K][D] fp32
  int i = blockIdx.x * blockDim.x + threadIdx.x;
  if (i >= K_DIM * D_DIM) return;
  int k = i >> 9, d = i & (D_DIM - 1);
  float v = emb[(size_t)d * K_DIM + k];
  embT[i] = v;
  float nr = norms[k]; nr = nr > 1e-12f ? nr : 1e-12f;
  ent[i] = (bf16_t)(v / nr);
}

__global__ void vq_transpose_bf16(const float* __restrict__ W, bf16_t* __restrict__ Wt,
                                  int rows, int cols) {  // W[rows][cols] -> Wt[cols][rows]
  int i = blockIdx.x * blockDim.x + threadIdx.x;
  if (i >= rows * cols) return;
  int wc = i / rows, wr = i - wc * rows;
  Wt[i] = (bf16_t)W[(size_t)wr * cols + wc];
}

// ------------------------------- output kernels ----------------------------
__global__ void vq_zero_f4(float4* __restrict__ p, size_t n4) {
  for (size_t i = (size_t)blockIdx.x * blockDim.x + threadIdx.x; i < n4;
       i += (size_t)gridDim.x * blockDim.x)
    p[i] = make_float4(0.f, 0.f, 0.f, 0.f);
}

__global__ void vq_scatter(const int* __restrict__ idx, float* __restrict__ enc,
                           float* __restrict__ encIdxOut) {
  int n = blockIdx.x * blockDim.x + threadIdx.x;
  if (n >= NTOK) return;
  int k = idx[n];
  enc[(size_t)n * K_DIM + k] = 1.0f;
  encIdxOut[n] = (float)k;
}

__global__ void vq_gather_rows(const int* __restrict__ idx,
                               const float* __restrict__ codeout,  // [K][IN]
                               float* __restrict__ qout) {         // [N][IN]
  int n = blockIdx.x;
  int k = idx[n];
  const float4* src = (const float4*)(codeout + (size_t)k * IN_DIM);
  float4* dst = (float4*)(qout + (size_t)n * IN_DIM);
  dst[threadIdx.x] = src[threadIdx.x];
}

#define LOSS_BLOCKS 1024
__global__ __launch_bounds__(256)
void vq_loss_partial(const int* __restrict__ idx, const float* __restrict__ x,
                     const float* __restrict__ embT, float* __restrict__ part) {
  __shared__ float red[256];
  const size_t total = (size_t)NTOK * D_DIM;
  float s = 0.f;
  for (size_t e = (size_t)blockIdx.x * blockDim.x + threadIdx.x; e < total;
       e += (size_t)gridDim.x * blockDim.x) {
    int n = (int)(e >> 9), d = (int)(e & (D_DIM - 1));
    float q = embT[(size_t)idx[n] * D_DIM + d];
    float diff = q - x[e];
    s += diff * diff;
  }
  red[threadIdx.x] = s;
  __syncthreads();
  for (int o = 128; o > 0; o >>= 1) {
    if (threadIdx.x < o) red[threadIdx.x] += red[threadIdx.x + o];
    __syncthreads();
  }
  if (threadIdx.x == 0) part[blockIdx.x] = red[0];
}

__global__ void vq_loss_final(const float* __restrict__ part, float* __restrict__ out) {
  float s = 0.f;
  for (int i = 0; i < LOSS_BLOCKS; ++i) s += part[i];   // deterministic order
  out[0] = 2.0f * s / ((float)NTOK * (float)D_DIM);     // commit + codebook
}

// --------------------------------- launcher --------------------------------
static inline size_t alignUp(size_t v) { return (v + 255) & ~(size_t)255; }

extern "C" void kernel_launch(void* const* d_in, const int* in_sizes, int n_in,
                              void* d_out, int out_size, void* d_ws, size_t ws_size,
                              hipStream_t stream) {
  (void)in_sizes; (void)n_in; (void)out_size; (void)ws_size;
  const float* inputs = (const float*)d_in[0];   // [B,T,IN]
  const float* W_in   = (const float*)d_in[1];   // [IN,D]
  const float* b_in   = (const float*)d_in[2];   // [D]
  const float* emb    = (const float*)d_in[3];   // [D,K]
  const float* W_out  = (const float*)d_in[4];   // [D,IN]
  const float* b_out  = (const float*)d_in[5];   // [IN]

  // workspace carve-up
  char* ws = (char*)d_ws;
  size_t off = 0;
  float*  X      = (float*)(ws + off); off = alignUp(off + (size_t)NTOK * D_DIM * 4);
  bf16_t* WINT   = (bf16_t*)(ws + off); off = alignUp(off + (size_t)D_DIM * IN_DIM * 2);
  bf16_t* ENT    = (bf16_t*)(ws + off); off = alignUp(off + (size_t)K_DIM * D_DIM * 2);
  float*  EMBT   = (float*)(ws + off); off = alignUp(off + (size_t)K_DIM * D_DIM * 4);
  bf16_t* WOUTT  = (bf16_t*)(ws + off); off = alignUp(off + (size_t)IN_DIM * D_DIM * 2);
  float*  NORMS  = (float*)(ws + off); off = alignUp(off + (size_t)K_DIM * 4);
  float*  CODEO  = (float*)(ws + off); off = alignUp(off + (size_t)K_DIM * IN_DIM * 4);
  int*    IDX    = (int*)(ws + off); off = alignUp(off + (size_t)NTOK * 4);
  float*  PART   = (float*)(ws + off); off = alignUp(off + (size_t)LOSS_BLOCKS * 4);

  // output carve-up (return order: qout, encodings, indices, loss)
  float* OUT_Q    = (float*)d_out;
  float* OUT_ENC  = OUT_Q + (size_t)NTOK * IN_DIM;
  float* OUT_IDX  = OUT_ENC + (size_t)NTOK * K_DIM;
  float* OUT_LOSS = OUT_IDX + (size_t)NTOK;

  // prep
  vq_colnorm<<<(K_DIM + 255) / 256, 256, 0, stream>>>(emb, NORMS);
  vq_prep_codebook<<<(K_DIM * D_DIM) / 256, 256, 0, stream>>>(emb, NORMS, ENT, EMBT);
  vq_transpose_bf16<<<(IN_DIM * D_DIM) / 256, 256, 0, stream>>>(W_in, WINT, IN_DIM, D_DIM);
  vq_transpose_bf16<<<(D_DIM * IN_DIM) / 256, 256, 0, stream>>>(W_out, WOUTT, D_DIM, IN_DIM);

  // GEMM1: x = inputs @ W_in + b_in
  vq_gemm_bf16_store<<<dim3(NTOK / BM, D_DIM / BN), 256, 0, stream>>>(
      inputs, WINT, b_in, X, NTOK, D_DIM, IN_DIM);
  // GEMM3': codebook_out = embT @ W_out + b_out  (tiny — then gather)
  vq_gemm_bf16_store<<<dim3(K_DIM / BM, IN_DIM / BN), 256, 0, stream>>>(
      EMBT, WOUTT, b_out, CODEO, K_DIM, IN_DIM, D_DIM);
  // GEMM2 + argmax: idx = argmax_k x @ en_norm^T
  vq_gemm_argmax<<<NTOK / BM, 256, 0, stream>>>(X, ENT, IDX);

  // encodings: zero then scatter one-hot; indices as float
  vq_zero_f4<<<2048, 256, 0, stream>>>((float4*)OUT_ENC, (size_t)NTOK * K_DIM / 4);
  vq_scatter<<<NTOK / 256, 256, 0, stream>>>(IDX, OUT_ENC, OUT_IDX);
  // quantized_out = codebook_out[idx]
  vq_gather_rows<<<NTOK, 256, 0, stream>>>(IDX, CODEO, OUT_Q);
  // loss (deterministic two-pass)
  vq_loss_partial<<<LOSS_BLOCKS, 256, 0, stream>>>(IDX, X, EMBT, PART);
  vq_loss_final<<<1, 1, 0, stream>>>(PART, OUT_LOSS);
}
